// TriLSTM_37460704755940
// MI455X (gfx1250) — matrix operand order repaired and checked by reading, not compiled, not run on hardware
//
#include <hip/hip_runtime.h>
#include <hip/hip_bf16.h>

typedef __attribute__((ext_vector_type(16))) __bf16        v16bf;
typedef __attribute__((ext_vector_type(8)))  float         v8f;
typedef __attribute__((ext_vector_type(4)))  unsigned int  v4u;

#define GEMM_TM 64
#define GEMM_TN 64
#define GEMM_TK 32
#define LDS_PAD 40   // bf16 elements per LDS row (80 bytes: 16B-aligned rows, bank-disjoint frags)

__device__ __forceinline__ unsigned short f2bf(float f) {
  unsigned int u = __float_as_uint(f);
  u += 0x7FFFu + ((u >> 16) & 1u);          // round-to-nearest-even
  return (unsigned short)(u >> 16);
}

__device__ __forceinline__ unsigned long long pack4bf(float a, float b, float c, float d) {
  return (unsigned long long)f2bf(a)
       | ((unsigned long long)f2bf(b) << 16)
       | ((unsigned long long)f2bf(c) << 32)
       | ((unsigned long long)f2bf(d) << 48);
}

union Frag { struct { v4u lo, hi; } u; v16bf v; };

// C[M,N] = (acc ? C : 0) + A[M,K](f32) @ Wb[N,K](bf16)^T + bias[N]
// Double-buffered: async global->LDS W staging (ASYNCcnt) of tile k+1 overlaps WMMA of tile k.
// FULL=true: all tiles 64-aligned -> branch-free hot loop (host dispatches).
template <bool FULL>
__global__ __launch_bounds__(128) void gemm_bf16w_kernel(
    float* __restrict__ C, const float* __restrict__ A, const unsigned short* __restrict__ Wb,
    const float* __restrict__ bias, int M, int N, int K, int accumulate)
{
  __shared__ __attribute__((aligned(16))) unsigned short sA[2][GEMM_TM * LDS_PAD];
  __shared__ __attribute__((aligned(16))) unsigned short sW[2][GEMM_TN * LDS_PAD];

  const int tid  = threadIdx.x;
  const int wave = tid >> 5;      // 0..3 : owns M rows [16*wave, 16*wave+16)
  const int lane = tid & 31;
  const int g    = lane >> 4;
  const int ln   = lane & 15;
  const int m0   = blockIdx.y * GEMM_TM;
  const int n0   = blockIdx.x * GEMM_TN;

  v8f acc[4];
  for (int i = 0; i < 4; ++i)
    for (int j = 0; j < 8; ++j) acc[i][j] = 0.0f;

  auto stageW = [&](int k0, unsigned short* dstW) {
#pragma unroll
    for (int i = 0; i < 2; ++i) {
      int idx = tid + i * 128;                 // 256 chunks of 8 bf16 (16B)
      int r = idx >> 2, c8 = (idx & 3) * 8;
      if (FULL) {
        unsigned loff = (unsigned)(size_t)&dstW[r * LDS_PAD + c8];
        const unsigned short* gp = &Wb[(long)(n0 + r) * K + (k0 + c8)];
        asm volatile("global_load_async_to_lds_b128 %0, %1, off"
                     :: "v"(loff), "v"((unsigned long long)(size_t)gp) : "memory");
      } else {
        int ng = n0 + r;
        unsigned long long lo = 0ull, hi = 0ull;
        if (ng < N) {
          const unsigned long long* gp = (const unsigned long long*)&Wb[(long)ng * K + (k0 + c8)];
          lo = gp[0]; hi = gp[1];
        }
        unsigned long long* sp = (unsigned long long*)&dstW[r * LDS_PAD + c8];
        sp[0] = lo; sp[1] = hi;
      }
    }
  };

  auto stageA = [&](int k0, unsigned short* dstA) {
#pragma unroll
    for (int i = 0; i < 4; ++i) {
      int idx = tid + i * 128;                 // 512 chunks of 4 f32
      int r = idx >> 3, c4 = (idx & 7) * 4;
      float4 v;
      if (FULL) {
        v = *(const float4*)&A[(long)(m0 + r) * K + (k0 + c4)];
      } else {
        v = make_float4(0.f, 0.f, 0.f, 0.f);
        if (m0 + r < M) v = *(const float4*)&A[(long)(m0 + r) * K + (k0 + c4)];
      }
      *(unsigned long long*)&dstA[r * LDS_PAD + c4] = pack4bf(v.x, v.y, v.z, v.w);
    }
  };

  const int nk = K >> 5;                       // K is a multiple of 32 in this model
  stageW(0, sW[0]);
  stageA(0, sA[0]);

  for (int it = 0; it < nk; ++it) {
    // Retire stage(it) (async W + LDS A stores) and guarantee buffer (it&1) reads from
    // iteration it-2 are finished workgroup-wide before it is overwritten next.
    asm volatile("s_wait_asynccnt 0x0" ::: "memory");
    __syncthreads();

    if (it + 1 < nk) {                         // prefetch next tile into the other buffer
      stageW((it + 1) << 5, sW[(it + 1) & 1]);
      stageA((it + 1) << 5, sA[(it + 1) & 1]);
    }

    const unsigned short* cA = sA[it & 1];
    const unsigned short* cW = sW[it & 1];

    // A fragment (16x32): lane ln = row m; VGPR0-3: K=8g+0..7, VGPR4-7: K=16+8g+0..7
    const int arow = wave * 16 + ln;
    Frag fa;
    fa.u.lo = *(const v4u*)&cA[arow * LDS_PAD + 8 * g];
    fa.u.hi = *(const v4u*)&cA[arow * LDS_PAD + 16 + 8 * g];

#pragma unroll
    for (int nt = 0; nt < 4; ++nt) {
      // B fragment (32x16): lane ln = col n; VGPR0-3: K=8g+0..7, VGPR4-7: K=16+8g+0..7
      const int brow = nt * 16 + ln;
      Frag fb;
      fb.u.lo = *(const v4u*)&cW[brow * LDS_PAD + 8 * g];
      fb.u.hi = *(const v4u*)&cW[brow * LDS_PAD + 16 + 8 * g];
      acc[nt] = __builtin_amdgcn_wmma_f32_16x16x32_bf16(
          false, fa.v, false, fb.v, (short)0, acc[nt], false, false);
    }
  }

  // Epilogue: f32 16x16 C/D layout -> lane (g,ln), VGPR r = element (m = 16*wave + r + 8g, n = ln)
  for (int nt = 0; nt < 4; ++nt) {
    int ng = n0 + nt * 16 + ln;
    if (!FULL && ng >= N) continue;
    float bv = bias ? bias[ng] : 0.0f;
    for (int r = 0; r < 8; ++r) {
      int mg = m0 + wave * 16 + r + 8 * g;
      if (FULL || mg < M) {
        long off = (long)mg * N + ng;
        float v = acc[nt][r] + bv;
        if (accumulate) v += C[off];
        C[off] = v;
      }
    }
  }
}

// ---------------- pointwise / helper kernels ----------------

__global__ void f32_to_bf16_kernel(unsigned short* __restrict__ dst,
                                   const float* __restrict__ src, int n)
{
  int id = (blockIdx.x * blockDim.x + threadIdx.x) * 4;
  if (id >= n) return;
  if (id + 3 < n) {
    const float4 v = *(const float4*)&src[id];
    *(unsigned long long*)&dst[id] = pack4bf(v.x, v.y, v.z, v.w);
  } else {
    for (int i = id; i < n; ++i) dst[i] = f2bf(src[i]);
  }
}

__global__ void extract_channel_kernel(float* __restrict__ out, const float* __restrict__ enc,
                                       int ch, int B, int N, int D)
{
  int id = blockIdx.x * blockDim.x + threadIdx.x;
  int nd = N * D;
  if (id >= B * nd) return;
  int b = id / nd, r = id % nd;
  out[id] = enc[(long)b * 2 * nd + (long)ch * nd + r];
}

__global__ void mean_feats_kernel(float* __restrict__ out, const float* __restrict__ feats,
                                  int B, int N, int D)
{
  int id = blockIdx.x * blockDim.x + threadIdx.x;
  if (id >= B * D) return;
  int b = id / D, d = id % D;
  const float* f = feats + (long)b * N * D + d;
  float s = 0.0f;
  for (int n = 0; n < N; ++n) s += f[(long)n * D];
  out[id] = s / (float)N;
}

__global__ void build_xlang_kernel(float* __restrict__ X, const int* __restrict__ seq, int t, int T,
                                   const float* __restrict__ W_emb, const float* __restrict__ fcf,
                                   const float* __restrict__ h1, const float* __restrict__ h2,
                                   int B, int D)
{
  int id = blockIdx.x * blockDim.x + threadIdx.x;
  int w = 4 * D;
  if (id >= B * w) return;
  int b = id / w, r = id % w, s = r / D, d = r % D;
  float v;
  if (s == 0) { int tok = seq[b * T + t]; v = W_emb[(long)tok * D + d]; }
  else if (s == 1) v = fcf[b * D + d];
  else if (s == 2) v = h1[b * D + d];
  else             v = h2[b * D + d];
  X[id] = v;
}

__global__ void concat3_kernel(float* __restrict__ X, const float* __restrict__ s0,
                               const float* __restrict__ s1, const float* __restrict__ s2,
                               int ns, int B, int D)
{
  int id = blockIdx.x * blockDim.x + threadIdx.x;
  int w = ns * D;
  if (id >= B * w) return;
  int b = id / w, r = id % w, s = r / D, d = r % D;
  const float* src = (s == 0) ? s0 : ((s == 1) ? s1 : s2);
  X[id] = src[b * D + d];
}

__device__ __forceinline__ float sigmoidf(float x) { return 1.0f / (1.0f + expf(-x)); }

__global__ void lstm_pw_kernel(const float* __restrict__ gates, const float* __restrict__ c_prev,
                               float* __restrict__ h_out, float* __restrict__ c_out, int B, int D)
{
  int id = blockIdx.x * blockDim.x + threadIdx.x;
  if (id >= B * D) return;
  int b = id / D, d = id % D;
  const float* gb = gates + (long)b * 4 * D;
  float i  = sigmoidf(gb[d]);
  float f  = sigmoidf(gb[D + d]);
  float gg = tanhf(gb[2 * D + d]);
  float o  = sigmoidf(gb[3 * D + d]);
  float c2 = f * c_prev[id] + i * gg;
  c_out[id] = c2;
  h_out[id] = o * tanhf(c2);
}

// dot[b,n] = ba + sum_d tanh(P[b,n,d] + ah[b,d]) * wa[d]
__global__ __launch_bounds__(128) void att_dot_kernel(
    const float* __restrict__ P, const float* __restrict__ ah,
    const float* __restrict__ wa, const float* __restrict__ ba,
    float* __restrict__ dot, int N, int D)
{
  int bn = blockIdx.x;
  int b = bn / N, n = bn % N;
  const float* p = P + ((long)b * N + n) * D;
  const float* a = ah + (long)b * D;
  float s = 0.0f;
  for (int d = threadIdx.x; d < D; d += 128) s += tanhf(p[d] + a[d]) * wa[d];
  __shared__ float red[128];
  red[threadIdx.x] = s;
  __syncthreads();
  for (int off = 64; off > 0; off >>= 1) {
    if (threadIdx.x < off) red[threadIdx.x] += red[threadIdx.x + off];
    __syncthreads();
  }
  if (threadIdx.x == 0) dot[bn] = red[0] + ba[0];
}

// softmax over n, mask, renormalize, then out[b,d] = sum_n w[n] * feats[b,n,d]
__global__ __launch_bounds__(256) void att_softmax_ws_kernel(
    const float* __restrict__ dot, const float* __restrict__ mask,
    const float* __restrict__ feats, float* __restrict__ out, int N, int D)
{
  int b = blockIdx.x;
  __shared__ float wsh[64];
  const float* db = dot + (long)b * N;
  const float* mb = mask + (long)b * N;
  if (threadIdx.x == 0) {
    float mx = -1e30f;
    for (int n = 0; n < N; ++n) mx = fmaxf(mx, db[n]);
    float s = 0.0f;
    for (int n = 0; n < N; ++n) { float e = expf(db[n] - mx); wsh[n] = e; s += e; }
    float s2 = 0.0f;
    for (int n = 0; n < N; ++n) { wsh[n] = wsh[n] / s * mb[n]; s2 += wsh[n]; }
    float inv = 1.0f / s2;
    for (int n = 0; n < N; ++n) wsh[n] *= inv;
  }
  __syncthreads();
  const float* fb = feats + (long)b * N * D;
  for (int d = threadIdx.x; d < D; d += 256) {
    float acc = 0.0f;
    for (int n = 0; n < N; ++n) acc += wsh[n] * fb[(long)n * D + d];
    out[(long)b * D + d] = acc;
  }
}

__global__ __launch_bounds__(256) void logsoftmax_kernel(
    const float* __restrict__ logits, float* __restrict__ out, int V, int t, int T)
{
  int b = blockIdx.x;
  const float* lb = logits + (long)b * V;
  __shared__ float red[256];
  float mx = -1e30f;
  for (int v = threadIdx.x; v < V; v += 256) mx = fmaxf(mx, lb[v]);
  red[threadIdx.x] = mx;
  __syncthreads();
  for (int off = 128; off > 0; off >>= 1) {
    if (threadIdx.x < off) red[threadIdx.x] = fmaxf(red[threadIdx.x], red[threadIdx.x + off]);
    __syncthreads();
  }
  mx = red[0];
  __syncthreads();
  float s = 0.0f;
  for (int v = threadIdx.x; v < V; v += 256) s += expf(lb[v] - mx);
  red[threadIdx.x] = s;
  __syncthreads();
  for (int off = 128; off > 0; off >>= 1) {
    if (threadIdx.x < off) red[threadIdx.x] += red[threadIdx.x + off];
    __syncthreads();
  }
  float lse = mx + logf(red[0]);
  float* ob = out + ((long)b * T + t) * V;
  for (int v = threadIdx.x; v < V; v += 256) ob[v] = lb[v] - lse;
}

// ---------------- host orchestration ----------------

static inline void gemm(float* C, const float* A, const unsigned short* Wb, const float* bias,
                        int M, int N, int K, int acc, hipStream_t s)
{
  dim3 grid((N + GEMM_TN - 1) / GEMM_TN, (M + GEMM_TM - 1) / GEMM_TM);
  if ((M % GEMM_TM == 0) && (N % GEMM_TN == 0))
    gemm_bf16w_kernel<true><<<grid, 128, 0, s>>>(C, A, Wb, bias, M, N, K, acc);
  else
    gemm_bf16w_kernel<false><<<grid, 128, 0, s>>>(C, A, Wb, bias, M, N, K, acc);
}

#define CEILDIV(a, b) (((a) + (b) - 1) / (b))

extern "C" void kernel_launch(void* const* d_in, const int* in_sizes, int n_in,
                              void* d_out, int out_size, void* d_ws, size_t ws_size,
                              hipStream_t stream)
{
  const int D = 512;
  const int B = 128;
  const int V = in_sizes[5];
  const int N = in_sizes[2] / B;
  const int T = in_sizes[0] / B;

  const int*   seq      = (const int*)  d_in[0];
  const float* enc      = (const float*)d_in[1];
  const float* mask     = (const float*)d_in[2];
  const float* W_emb    = (const float*)d_in[3];
  const float* fc_W     = (const float*)d_in[4];
  const float* fc_b     = (const float*)d_in[5];
  const float* lang_Wih = (const float*)d_in[6];
  const float* lang_Whh = (const float*)d_in[7];
  const float* lang_bih = (const float*)d_in[8];
  const float* lang_bhh = (const float*)d_in[9];
  const float* att_Wih  = (const float*)d_in[10];
  const float* att_Whh  = (const float*)d_in[11];
  const float* att_bih  = (const float*)d_in[12];
  const float* att_bhh  = (const float*)d_in[13];
  const float* gcn_Wih  = (const float*)d_in[14];
  const float* gcn_Whh  = (const float*)d_in[15];
  const float* gcn_bih  = (const float*)d_in[16];
  const float* gcn_bhh  = (const float*)d_in[17];
  const float* top_Wp   = (const float*)d_in[18];
  const float* top_bp   = (const float*)d_in[19];
  const float* top_Wh   = (const float*)d_in[20];
  const float* top_bh   = (const float*)d_in[21];
  const float* top_wa   = (const float*)d_in[22];
  const float* top_ba   = (const float*)d_in[23];
  const float* mid_Wp   = (const float*)d_in[24];
  const float* mid_bp   = (const float*)d_in[25];
  const float* mid_Wh   = (const float*)d_in[26];
  const float* mid_bh   = (const float*)d_in[27];
  const float* mid_wa   = (const float*)d_in[28];
  const float* mid_ba   = (const float*)d_in[29];

  char* ws = (char*)d_ws;
  size_t off = 0;
  auto alloc = [&](size_t bytes) -> void* {
    size_t a = (off + 255) & ~(size_t)255;
    off = a + bytes;
    return (void*)(ws + a);
  };

  const size_t BND = (size_t)B * N * D;
  float* base_c  = (float*)alloc(BND * 4);
  float* gcn_c   = (float*)alloc(BND * 4);
  float* P_base  = (float*)alloc(BND * 4);
  float* P_gcn   = (float*)alloc(BND * 4);
  float* fc_feats= (float*)alloc((size_t)B * D * 4);
  float* hbuf    = (float*)alloc((size_t)6 * B * D * 4);
  float* cbuf    = (float*)alloc((size_t)6 * B * D * 4);
  float* X       = (float*)alloc((size_t)B * 4 * D * 4);
  float* G       = (float*)alloc((size_t)B * 4 * D * 4);
  float* ah      = (float*)alloc((size_t)B * D * 4);
  float* dotb    = (float*)alloc((size_t)B * N * 4);
  float* attval  = (float*)alloc((size_t)B * D * 4);
  float* logits  = (float*)alloc((size_t)B * V * 4);
  (void)ws_size;

  // bf16 weight copies (time-invariant -> converted once per launch)
  const float* wsrc[11] = { lang_Wih, lang_Whh, att_Wih, att_Whh, gcn_Wih, gcn_Whh,
                            fc_W, top_Wp, top_Wh, mid_Wp, mid_Wh };
  const size_t wcnt[11] = { (size_t)4*D*4*D, (size_t)4*D*D, (size_t)4*D*3*D, (size_t)4*D*D,
                            (size_t)4*D*2*D, (size_t)4*D*D, (size_t)V*D,
                            (size_t)D*D, (size_t)D*D, (size_t)D*D, (size_t)D*D };
  unsigned short* bW[11];
  for (int i = 0; i < 11; ++i) {
    bW[i] = (unsigned short*)alloc(wcnt[i] * 2);
    f32_to_bf16_kernel<<<CEILDIV(wcnt[i] / 4, 256), 256, 0, stream>>>(bW[i], wsrc[i], (int)wcnt[i]);
  }
  unsigned short *b_lang_Wih = bW[0], *b_lang_Whh = bW[1], *b_att_Wih = bW[2], *b_att_Whh = bW[3],
                 *b_gcn_Wih = bW[4], *b_gcn_Whh = bW[5], *b_fc_W = bW[6],
                 *b_top_Wp = bW[7], *b_top_Wh = bW[8], *b_mid_Wp = bW[9], *b_mid_Wh = bW[10];

  // -------- time-invariant precompute --------
  extract_channel_kernel<<<CEILDIV(B * N * D, 256), 256, 0, stream>>>(base_c, enc, 0, B, N, D);
  extract_channel_kernel<<<CEILDIV(B * N * D, 256), 256, 0, stream>>>(gcn_c,  enc, 1, B, N, D);
  mean_feats_kernel<<<CEILDIV(B * D, 256), 256, 0, stream>>>(fc_feats, base_c, B, N, D);
  gemm(P_base, base_c, b_top_Wp, top_bp, B * N, D, D, 0, stream);
  gemm(P_gcn,  gcn_c,  b_mid_Wp, mid_bp, B * N, D, D, 0, stream);

  hipMemsetAsync(hbuf, 0, (size_t)6 * B * D * 4, stream);
  hipMemsetAsync(cbuf, 0, (size_t)6 * B * D * 4, stream);

  float* hset[2][3], *cset[2][3];
  for (int s = 0; s < 2; ++s)
    for (int i = 0; i < 3; ++i) {
      hset[s][i] = hbuf + ((size_t)(s * 3 + i)) * B * D;
      cset[s][i] = cbuf + ((size_t)(s * 3 + i)) * B * D;
    }

  int prev = 0, next = 1;
  const int ew = 256;

  for (int t = 0; t < T; ++t) {
    float **hp = hset[prev], **cp = cset[prev];
    float **hn = hset[next], **cn = cset[next];

    // ---- language LSTM: x = [emb_t | fc_feats | h_gcn_prev | h_att_prev]
    build_xlang_kernel<<<CEILDIV(B * 4 * D, ew), ew, 0, stream>>>(
        X, seq, t, T, W_emb, fc_feats, hp[1], hp[2], B, D);
    gemm(G, X,     b_lang_Wih, lang_bih, B, 4 * D, 4 * D, 0, stream);
    gemm(G, hp[0], b_lang_Whh, lang_bhh, B, 4 * D, D,     1, stream);
    lstm_pw_kernel<<<CEILDIV(B * D, ew), ew, 0, stream>>>(G, cp[0], hn[0], cn[0], B, D);

    // ---- top attention on base feats, query = h_lang
    gemm(ah, hn[0], b_top_Wh, top_bh, B, D, D, 0, stream);
    att_dot_kernel<<<B * N, 128, 0, stream>>>(P_base, ah, top_wa, top_ba, dotb, N, D);
    att_softmax_ws_kernel<<<B, 256, 0, stream>>>(dotb, mask, base_c, attval, N, D);

    // ---- attention LSTM: ctx = [base_att | h_lang | h_lang], state slot 2
    concat3_kernel<<<CEILDIV(B * 3 * D, ew), ew, 0, stream>>>(X, attval, hn[0], hn[0], 3, B, D);
    gemm(G, X,     b_att_Wih, att_bih, B, 4 * D, 3 * D, 0, stream);
    gemm(G, hp[2], b_att_Whh, att_bhh, B, 4 * D, D,     1, stream);
    lstm_pw_kernel<<<CEILDIV(B * D, ew), ew, 0, stream>>>(G, cp[2], hn[2], cn[2], B, D);

    // ---- mid attention on gcn feats, query = h_att
    gemm(ah, hn[2], b_mid_Wh, mid_bh, B, D, D, 0, stream);
    att_dot_kernel<<<B * N, 128, 0, stream>>>(P_gcn, ah, mid_wa, mid_ba, dotb, N, D);
    att_softmax_ws_kernel<<<B, 256, 0, stream>>>(dotb, mask, gcn_c, attval, N, D);

    // ---- gcn LSTM: in = [gcn_att | h_att], state slot 1
    concat3_kernel<<<CEILDIV(B * 2 * D, ew), ew, 0, stream>>>(X, attval, hn[2], hn[2], 2, B, D);
    gemm(G, X,     b_gcn_Wih, gcn_bih, B, 4 * D, 2 * D, 0, stream);
    gemm(G, hp[1], b_gcn_Whh, gcn_bhh, B, 4 * D, D,     1, stream);
    lstm_pw_kernel<<<CEILDIV(B * D, ew), ew, 0, stream>>>(G, cp[1], hn[1], cn[1], B, D);

    // ---- vocabulary projection from h_att + log_softmax into d_out[:, t, :]
    gemm(logits, hn[2], b_fc_W, fc_b, B, V, D, 0, stream);
    logsoftmax_kernel<<<B, 256, 0, stream>>>(logits, (float*)d_out, V, t, T);

    prev ^= 1; next ^= 1;
  }
}
